// GNN_73280732004501
// MI455X (gfx1250) — matrix operand order repaired and checked by reading, not compiled, not run on hardware
//
#include <hip/hip_runtime.h>
#include <hip/hip_bf16.h>

// ---------------------------------------------------------------------------
// GraphConv x3 for MI455X (gfx1250, wave32, WMMA, LDS-staged weights).
// Per layer:
//   zero(agg_f32) -> scatter_add (gather, scale, global_atomic_add_f32)
//   cvt agg -> bf16
//   GEMM: out = act( agg_bf @ Wrel^T + in_bf @ Wroot^T + b )
//         v_wmma_f32_16x16x32_bf16, B staged in LDS via async-to-LDS
// ---------------------------------------------------------------------------

typedef __attribute__((ext_vector_type(16))) __bf16    v16bf;
typedef __attribute__((ext_vector_type(8)))  float     v8f;
typedef __attribute__((ext_vector_type(4)))  unsigned  u32x4;
typedef __attribute__((ext_vector_type(4)))  int       i32x4;

struct Q2 { u32x4 a, b; };   // 32B = one v16bf fragment

#if defined(__HIP_DEVICE_COMPILE__) && __has_builtin(__builtin_amdgcn_global_load_async_to_lds_b128)
#define GNN_ASYNC_LDS 1
typedef __attribute__((address_space(1))) i32x4* gasync_ptr;
typedef __attribute__((address_space(3))) i32x4* lasync_ptr;
#endif

// ---------------------------------------------------------------------------
// helpers
static __device__ __forceinline__ unsigned pack_bf2(float lo, float hi) {
    unsigned a = __builtin_bit_cast(unsigned short, (__bf16)lo);
    unsigned b = __builtin_bit_cast(unsigned short, (__bf16)hi);
    return a | (b << 16);   // fuses to v_cvt_pk_bf16_f32
}
static __device__ __forceinline__ float bfbits2f(unsigned bits16) {
    return __builtin_bit_cast(float, bits16 << 16);
}
// A fragment from bf16 row-major memory (16-bit A 16x32 lane layout):
// lane holds K[ka..ka+7] and K[ka+16..ka+23] (p already lane-adjusted).
static __device__ __forceinline__ v16bf load_a_bf(const __bf16* __restrict__ p) {
    Q2 t;
    t.a = *reinterpret_cast<const u32x4*>(p);
    t.b = *reinterpret_cast<const u32x4*>(p + 16);
    return __builtin_bit_cast(v16bf, t);
}
static __device__ __forceinline__ v16bf lds_b_frag(const u32x4* sB, int chunk) {
    Q2 t;
    t.a = sB[2 * chunk + 0];
    t.b = sB[2 * chunk + 1];
    return __builtin_bit_cast(v16bf, t);
}

// ---------------------------------------------------------------------------
__global__ __launch_bounds__(256) void gnn_zero_f4(float4* __restrict__ p,
                                                   long long n4) {
    long long i      = (long long)blockIdx.x * blockDim.x + threadIdx.x;
    long long stride = (long long)gridDim.x * blockDim.x;
    float4 z = make_float4(0.f, 0.f, 0.f, 0.f);
    for (; i < n4; i += stride) p[i] = z;
}

// f32 -> bf16, 8 elements / thread (2x b128 load, 1x b128 store)
__global__ __launch_bounds__(256) void gnn_cvt_bf16(const float* __restrict__ in,
                                                    unsigned* __restrict__ out,
                                                    long long n8) {
    long long t = (long long)blockIdx.x * blockDim.x + threadIdx.x;
    if (t >= n8) return;
    const float4* p = reinterpret_cast<const float4*>(in) + 2 * t;
    float4 f0 = p[0], f1 = p[1];
    u32x4 q;
    q[0] = pack_bf2(f0.x, f0.y);
    q[1] = pack_bf2(f0.z, f0.w);
    q[2] = pack_bf2(f1.x, f1.y);
    q[3] = pack_bf2(f1.z, f1.w);
    reinterpret_cast<u32x4*>(out)[t] = q;
}

// agg[dst,:] += w * x[src,:]   (f32 gather, layer 1)
__global__ __launch_bounds__(256) void gnn_scatter_f32(
        const float* __restrict__ x, const int* __restrict__ src,
        const int* __restrict__ dst, const float* __restrict__ w,
        float* __restrict__ agg, int E, int D4) {
    long long tid   = (long long)blockIdx.x * blockDim.x + threadIdx.x;
    long long total = (long long)E * D4;
    if (tid >= total) return;
    int e = (int)(tid / D4), f = (int)(tid % D4);
    int s = src[e], d = dst[e];
    float we = w[e];
    float4 v = reinterpret_cast<const float4*>(x + (size_t)s * (D4 * 4))[f];
    float* o = agg + (size_t)d * (D4 * 4) + (size_t)f * 4;
    atomicAdd(o + 0, we * v.x);
    atomicAdd(o + 1, we * v.y);
    atomicAdd(o + 2, we * v.z);
    atomicAdd(o + 3, we * v.w);
}

// agg[dst,:] += w * x[src,:]   (bf16 gather, layers 2-3)
__global__ __launch_bounds__(256) void gnn_scatter_bf16(
        const __bf16* __restrict__ x, const int* __restrict__ src,
        const int* __restrict__ dst, const float* __restrict__ w,
        float* __restrict__ agg, int E, int D4) {
    long long tid   = (long long)blockIdx.x * blockDim.x + threadIdx.x;
    long long total = (long long)E * D4;
    if (tid >= total) return;
    int e = (int)(tid / D4), f = (int)(tid % D4);
    int s = src[e], d = dst[e];
    float we = w[e];
    uint2 v = reinterpret_cast<const uint2*>(x + (size_t)s * (D4 * 4))[f];
    float* o = agg + (size_t)d * (D4 * 4) + (size_t)f * 4;
    atomicAdd(o + 0, we * bfbits2f(v.x & 0xffffu));
    atomicAdd(o + 1, we * bfbits2f(v.x >> 16));
    atomicAdd(o + 2, we * bfbits2f(v.y & 0xffffu));
    atomicAdd(o + 3, we * bfbits2f(v.y >> 16));
}

// ---------------------------------------------------------------------------
// Fused dual GEMM. Block = 8 waves = 8 row tiles (128 rows) x (COLS*16) cols.
// B (both weight matrices) staged in LDS once per K-step for the whole block.
template <int COLS, bool RELU, bool OUTBF>
__global__ __launch_bounds__(256) void gnn_gemm_fused(
        const __bf16* __restrict__ A0, const __bf16* __restrict__ A1,
        const __bf16* __restrict__ W0, const __bf16* __restrict__ W1,
        const float* __restrict__ bias, float* __restrict__ outf,
        __bf16* __restrict__ outb, int nrows, int K, int Hout) {
    constexpr int CG = COLS * 16;          // columns per block
    __shared__ u32x4 sB[CG * 8];           // 2 matrices * CG cols * 64B = CG*128 B

    const int lane    = threadIdx.x & 31;
    const int wavblk  = threadIdx.x >> 5;
    const int ln16    = lane & 15;
    const bool hi     = lane >= 16;

    const int rowTiles  = nrows >> 4;
    const int rowGroups = (rowTiles + 7) >> 3;
    const int rowGrp    = blockIdx.x % rowGroups;
    const int colBase   = (blockIdx.x / rowGroups) * CG;

    int rowTile = rowGrp * 8 + wavblk;
    const bool rowValid = rowTile < rowTiles;
    if (!rowValid) rowTile = rowTiles - 1;      // clamp: keep EXEC all-1s
    const int r0 = rowTile << 4;

    const __bf16* a0p = A0 + (size_t)(r0 + ln16) * K + (hi ? 8 : 0);
    const __bf16* a1p = A1 + (size_t)(r0 + ln16) * K + (hi ? 8 : 0);

    v8f acc[COLS];
#pragma unroll
    for (int c = 0; c < COLS; ++c) acc[c] = v8f{0.f,0.f,0.f,0.f,0.f,0.f,0.f,0.f};

    for (int k0 = 0; k0 < K; k0 += 32) {
        // ---- stage B tiles (both matrices) into LDS: CG*8 16-byte units ----
        for (int u = threadIdx.x; u < CG * 8; u += 256) {
            int half16 = u & 1;
            int chunk  = u >> 1;               // ((m*CG + c)*2 + khalf)
            int m      = chunk / (CG * 2);
            int rem    = chunk - m * (CG * 2);
            int c      = rem >> 1;
            int kh     = rem & 1;
            int col    = colBase + c;
            if (col >= Hout) col = Hout - 1;   // clamp (stores are masked)
            const __bf16* wsrc = (m ? W1 : W0) +
                                 (size_t)col * K + k0 + kh * 8 + half16 * 16;
#ifdef GNN_ASYNC_LDS
            __builtin_amdgcn_global_load_async_to_lds_b128(
                (gasync_ptr)wsrc, (lasync_ptr)&sB[u], 0, 0);
#else
            sB[u] = *reinterpret_cast<const u32x4*>(wsrc);
#endif
        }
#ifdef GNN_ASYNC_LDS
        asm volatile("s_wait_asynccnt 0" ::: "memory");
#endif
        __syncthreads();

        // ---- A fragments (shared by all COLS column tiles) ----
        v16bf a0 = load_a_bf(a0p + k0);
        v16bf a1 = load_a_bf(a1p + k0);

        // ---- 2*COLS WMMAs reusing a0/a1 ----
#pragma unroll
        for (int c = 0; c < COLS; ++c) {
            int bcol = c * 16 + ln16;
            v16bf b0 = lds_b_frag(sB, (0 * CG + bcol) * 2 + (hi ? 1 : 0));
            acc[c] = __builtin_amdgcn_wmma_f32_16x16x32_bf16(
                         false, a0, false, b0, (short)0, acc[c], false, false);
            v16bf b1 = lds_b_frag(sB, (1 * CG + bcol) * 2 + (hi ? 1 : 0));
            acc[c] = __builtin_amdgcn_wmma_f32_16x16x32_bf16(
                         false, a1, false, b1, (short)0, acc[c], false, false);
        }
        __syncthreads();
    }

    // ---- epilogue: bias + relu, store (f32 or bf16) ----
    const int rbase = r0 + (hi ? 8 : 0);
#pragma unroll
    for (int c = 0; c < COLS; ++c) {
        int col = colBase + c * 16 + ln16;
        if (rowValid && col < Hout) {
            float bv = bias[col];
#pragma unroll
            for (int r = 0; r < 8; ++r) {
                float v = acc[c][r] + bv;
                if (RELU) v = fmaxf(v, 0.f);
                size_t idx = (size_t)(rbase + r) * Hout + col;
                if (OUTBF) outb[idx] = (__bf16)v;
                else       outf[idx] = v;
            }
        }
    }
}

// ---------------------------------------------------------------------------
extern "C" void kernel_launch(void* const* d_in, const int* in_sizes, int n_in,
                              void* d_out, int out_size, void* d_ws, size_t ws_size,
                              hipStream_t stream) {
    const float* x   = (const float*)d_in[0];
    const int*   ei  = (const int*)  d_in[1];
    const float* ew  = (const float*)d_in[2];
    const float* W1r = (const float*)d_in[4];
    const float* b1  = (const float*)d_in[5];
    const float* W1o = (const float*)d_in[6];
    const float* W2r = (const float*)d_in[7];
    const float* b2  = (const float*)d_in[8];
    const float* W2o = (const float*)d_in[9];
    const float* W3r = (const float*)d_in[10];
    const float* b3  = (const float*)d_in[11];
    const float* W3o = (const float*)d_in[12];
    float* out = (float*)d_out;

    const int N = in_sizes[0] / 1152;
    const int E = in_sizes[2];
    const int* src = ei;
    const int* dst = ei + E;

    // Workspace overlay:
    //  [0, N*1152*4)              agg_f32 (all layers)  / x_bf16 after agg1 cvt
    //  [+0, +N*1152*2)            agg_bf16 (all layers)
    //  then h1_bf16 (N*256), h2_bf16 (N*128), weights_bf16 (~1.3MB)
    char* ws = (char*)d_ws;
    float*  aggf  = (float*) ws;
    __bf16* xbf   = (__bf16*)ws;                              // alias of aggf region
    __bf16* aggbf = (__bf16*)(ws + (size_t)N * 1152 * 4);
    __bf16* h1bf  = aggbf + (size_t)N * 1152;
    __bf16* h2bf  = h1bf  + (size_t)N * 256;
    __bf16* wpool = h2bf  + (size_t)N * 128;
    __bf16* w1r = wpool;
    __bf16* w1o = w1r + 256 * 1152;
    __bf16* w2r = w1o + 256 * 1152;
    __bf16* w2o = w2r + 128 * 256;
    __bf16* w3r = w2o + 128 * 256;
    __bf16* w3o = w3r + 2 * 128;

    const int B = 256;
    auto cvt = [&](const float* in, __bf16* o, long long n) {
        long long n8 = n / 8;
        gnn_cvt_bf16<<<(int)((n8 + B - 1) / B), B, 0, stream>>>(in, (unsigned*)o, n8);
    };

    // ---- weights -> bf16 (tiny) ----
    cvt(W1r, w1r, 256 * 1152);  cvt(W1o, w1o, 256 * 1152);
    cvt(W2r, w2r, 128 * 256);   cvt(W2o, w2o, 128 * 256);
    cvt(W3r, w3r, 2 * 128);     cvt(W3o, w3o, 2 * 128);

    // ---- Layer 1: 1152 -> 256, relu ----
    {
        gnn_zero_f4<<<4096, B, 0, stream>>>((float4*)aggf, (long long)N * 1152 / 4);
        long long t = (long long)E * (1152 / 4);
        gnn_scatter_f32<<<(int)((t + B - 1) / B), B, 0, stream>>>(
            x, src, dst, ew, aggf, E, 1152 / 4);
        cvt(aggf, aggbf, (long long)N * 1152);
        cvt(x, xbf, (long long)N * 1152);        // overwrites dead agg_f32 region
        int rowGroups = ((N / 16) + 7) / 8;
        gnn_gemm_fused<4, true, true><<<rowGroups * (256 / 64), B, 0, stream>>>(
            aggbf, xbf, w1r, w1o, b1, nullptr, h1bf, N, 1152, 256);
    }
    // ---- Layer 2: 256 -> 128, relu ----
    {
        gnn_zero_f4<<<4096, B, 0, stream>>>((float4*)aggf, (long long)N * 256 / 4);
        long long t = (long long)E * (256 / 4);
        gnn_scatter_bf16<<<(int)((t + B - 1) / B), B, 0, stream>>>(
            h1bf, src, dst, ew, aggf, E, 256 / 4);
        cvt(aggf, aggbf, (long long)N * 256);
        int rowGroups = ((N / 16) + 7) / 8;
        gnn_gemm_fused<4, true, true><<<rowGroups * (128 / 64), B, 0, stream>>>(
            aggbf, h1bf, w2r, w2o, b2, nullptr, h2bf, N, 256, 128);
    }
    // ---- Layer 3: 128 -> 2, no relu ----
    {
        gnn_zero_f4<<<4096, B, 0, stream>>>((float4*)aggf, (long long)N * 128 / 4);
        long long t = (long long)E * (128 / 4);
        gnn_scatter_bf16<<<(int)((t + B - 1) / B), B, 0, stream>>>(
            h2bf, src, dst, ew, aggf, E, 128 / 4);
        cvt(aggf, aggbf, (long long)N * 128);
        int rowGroups = ((N / 16) + 7) / 8;
        gnn_gemm_fused<1, false, false><<<rowGroups, B, 0, stream>>>(
            aggbf, h2bf, w3r, w3o, b3, out, nullptr, N, 128, 2);
    }
}
